// DenseGCN_24352464569903
// MI455X (gfx1250) — matrix operand order repaired and checked by reading, not compiled, not run on hardware
//
#include <hip/hip_runtime.h>
#include <hip/hip_bf16.h>

// ---------------------------------------------------------------------------
// DenseGCN / EdgeConv for MI455X (gfx1250), wave32 + WMMA bf16 16x16x32.
//   N=20000 nodes, E=320000 edges, growth 64, 3 blocks, concat width 256.
//   Feature buffer xbuf [N,256] f32 lives in workspace (20.48 MB, L2-resident).
//   msg GEMM: M=E, K=2C, N=64 -> v_wmma_f32_16x16x32_bf16.
//   B (weights) staged in LDS per workgroup; A gathered per edge as f32->bf16.
//   segment_max via atomicMax on uint bits (msgs are >= 0 after relu).
// ---------------------------------------------------------------------------

#define N_NODES 20000
#define N_EDGES 320000
#define XSTRIDE 256   // concat channel stride (f32)

typedef __attribute__((ext_vector_type(16))) __bf16 v16bf;
typedef __attribute__((ext_vector_type(8)))  float  v8f;

// ---- 1) init concat buffer: cols 0..63 = x, cols 64..255 = 0 --------------
__global__ void dgcn_init_xbuf(const float* __restrict__ x,
                               float* __restrict__ xbuf) {
    int t = blockIdx.x * blockDim.x + threadIdx.x;         // [0, N*256)
    if (t >= N_NODES * XSTRIDE) return;
    int n = t >> 8, c = t & 255;
    xbuf[t] = (c < 64) ? x[n * 64 + c] : 0.0f;
}

// ---- 2) pack W [2C,64] f32 -> bf16 in WMMA B-fragment order ---------------
// Fragment (kt, nt) covers K = kt*32..+31, N = nt*16..+15.
// Lane l, elem e (0..15): n = nt*16 + (l&15);
//   k = kt*32 + (l<16 ? 0 : 8) + (e<8 ? e : e+8)  (VGPR v holds K pair 2v,2v+1)
// Storage: packed[frag*512 + lane*16 + e]  (each lane's 16 bf16 contiguous).
__global__ void dgcn_pack_w(const float* __restrict__ W,
                            __bf16* __restrict__ packed, int twoC) {
    int t = blockIdx.x * blockDim.x + threadIdx.x;         // [0, 2C*64)
    if (t >= twoC * 64) return;
    int frag = t >> 9;
    int lane = (t >> 4) & 31;
    int e    = t & 15;
    int kt = frag >> 2, nt = frag & 3;
    int n = nt * 16 + (lane & 15);
    int k = kt * 32 + ((lane < 16) ? 0 : 8) + ((e < 8) ? e : e + 8);
    packed[t] = (__bf16)W[k * 64 + n];
}

// ---- 3) EdgeConv: gather -> bf16 WMMA GEMM -> relu -> atomic segment-max --
// One wave handles 32 edges x 64 outputs: two A fragments, 8 accumulators.
// Weights live in LDS (max 48 KB), loaded once per workgroup.
template<int C>
__global__ __launch_bounds__(256)
void dgcn_edgeconv_wmma(float* xbuf,                        // [N,256] rw
                        const __bf16* __restrict__ packedW, // fragment-packed
                        const float* __restrict__ bias,     // [64]
                        const int* __restrict__ ei) {       // [2,E]
    constexpr int KSTEPS = (2 * C) / 32;                    // K tiles of 32
    constexpr int OUTCOL = C;         // block i writes cols C..C+63 (C=64(i+1))
    constexpr int WELEMS = 2 * C * 64;                      // bf16 elements

    __shared__ __bf16 sW[WELEMS];
    // Cooperative 16B-vector copy of packed weights into LDS.
    {
        const uint4* src = (const uint4*)packedW;
        uint4* dst = (uint4*)sW;
        for (int o = threadIdx.x; o < WELEMS / 8; o += 256) dst[o] = src[o];
    }
    __syncthreads();

    const int lane  = threadIdx.x & 31;
    const int wave  = blockIdx.x * (blockDim.x >> 5) + (threadIdx.x >> 5);
    const int ebase = wave * 32;                            // 32 edges per wave
    const int m     = lane & 15;
    const bool hi   = lane >= 16;

    // Per-lane edge rows for the two 16-row A tiles.
    const int e0 = ebase + m, e1 = e0 + 16;
    const float* xs0 = xbuf + (size_t)ei[e0]           * XSTRIDE;
    const float* xd0 = xbuf + (size_t)ei[N_EDGES + e0] * XSTRIDE;
    const float* xs1 = xbuf + (size_t)ei[e1]           * XSTRIDE;
    const float* xd1 = xbuf + (size_t)ei[N_EDGES + e1] * XSTRIDE;

    v8f acc0[4] = {}, acc1[4] = {};

    for (int kt = 0; kt < KSTEPS; ++kt) {
        // A fragment assembly per ISA layout: lane (lo/hi) K base +0/+8;
        // VGPR p: K = 2p,2p+1 (p<4) or 16+2(p-4).. (p>=4).
        const int kb = kt * 32 + (hi ? 8 : 0);
        v16bf a0, a1;
#pragma unroll
        for (int p = 0; p < 8; ++p) {
            const int k0 = kb + ((p < 4) ? (2 * p) : (16 + 2 * (p - 4)));
            float2 q0, q1;
            if (k0 < C) {                 // feat[:, 0:C] = x[dst]
                q0 = *(const float2*)(xd0 + k0);
                q1 = *(const float2*)(xd1 + k0);
            } else {                      // feat[:, C:2C] = x[src] - x[dst]
                const int c = k0 - C;
                const float2 s0 = *(const float2*)(xs0 + c);
                const float2 d0 = *(const float2*)(xd0 + c);
                const float2 s1 = *(const float2*)(xs1 + c);
                const float2 d1 = *(const float2*)(xd1 + c);
                q0.x = s0.x - d0.x;  q0.y = s0.y - d0.y;
                q1.x = s1.x - d1.x;  q1.y = s1.y - d1.y;
            }
            a0[2 * p] = (__bf16)q0.x;  a0[2 * p + 1] = (__bf16)q0.y;
            a1[2 * p] = (__bf16)q1.x;  a1[2 * p + 1] = (__bf16)q1.y;
        }
#pragma unroll
        for (int nt = 0; nt < 4; ++nt) {
            const v16bf b = *(const v16bf*)(sW + (((kt * 4 + nt) << 9) |
                                                  (lane << 4)));
            acc0[nt] = __builtin_amdgcn_wmma_f32_16x16x32_bf16(
                false, a0, false, b, (short)0, acc0[nt], false, false);
            acc1[nt] = __builtin_amdgcn_wmma_f32_16x16x32_bf16(
                false, a1, false, b, (short)0, acc1[nt], false, false);
        }
    }

    // Epilogue: +bias, relu, atomic segment-max into concat columns.
    // C/D layout: VGPR r -> row m = r + (hi?8:0), col n = lane&15.
    const int n0 = lane & 15;
    const int rb = hi ? 8 : 0;
#pragma unroll
    for (int nt = 0; nt < 4; ++nt) {
        const int n   = nt * 16 + n0;
        const float b = bias[n];
#pragma unroll
        for (int r = 0; r < 8; ++r) {
            const int row = rb + r;
            float v = fmaxf(acc0[nt][r] + b, 0.0f);
            int   d = ei[N_EDGES + ebase + row];
            atomicMax((unsigned int*)(xbuf + (size_t)d * XSTRIDE + OUTCOL + n),
                      __float_as_uint(v));      // valid: values are >= 0
            v = fmaxf(acc1[nt][r] + b, 0.0f);
            d = ei[N_EDGES + ebase + 16 + row];
            atomicMax((unsigned int*)(xbuf + (size_t)d * XSTRIDE + OUTCOL + n),
                      __float_as_uint(v));
        }
    }
}

// ---- 4) out[n,g] = max_j xbuf[n, 4g + j]  (row-major reshape (64,4)) ------
__global__ void dgcn_final_max(const float* __restrict__ xbuf,
                               float* __restrict__ out) {
    int t = blockIdx.x * blockDim.x + threadIdx.x;          // [0, N*64)
    if (t >= N_NODES * 64) return;
    int n = t >> 6, g = t & 63;
    float4 v = *(const float4*)(xbuf + (size_t)n * XSTRIDE + (g << 2));
    out[t] = fmaxf(fmaxf(v.x, v.y), fmaxf(v.z, v.w));
}

// ---------------------------------------------------------------------------
extern "C" void kernel_launch(void* const* d_in, const int* in_sizes, int n_in,
                              void* d_out, int out_size, void* d_ws,
                              size_t ws_size, hipStream_t stream) {
    (void)in_sizes; (void)n_in; (void)out_size; (void)ws_size;
    const float* x  = (const float*)d_in[0];
    const float* Ws[3] = {(const float*)d_in[1], (const float*)d_in[3],
                          (const float*)d_in[5]};
    const float* Bs[3] = {(const float*)d_in[2], (const float*)d_in[4],
                          (const float*)d_in[6]};
    const int* ei = (const int*)d_in[7];
    float* out = (float*)d_out;

    // Workspace: xbuf [N,256] f32, then bf16 fragment-packed weights.
    float*  xbuf = (float*)d_ws;
    __bf16* pw0  = (__bf16*)((char*)d_ws + (size_t)N_NODES * XSTRIDE * 4);
    __bf16* pw1  = pw0 + 128 * 64;
    __bf16* pw2  = pw1 + 256 * 64;

    dgcn_init_xbuf<<<(N_NODES * XSTRIDE) / 256, 256, 0, stream>>>(x, xbuf);
    dgcn_pack_w<<<(128 * 64) / 256, 256, 0, stream>>>(Ws[0], pw0, 128);
    dgcn_pack_w<<<(256 * 64) / 256, 256, 0, stream>>>(Ws[1], pw1, 256);
    dgcn_pack_w<<<(384 * 64) / 256, 256, 0, stream>>>(Ws[2], pw2, 384);

    // 8 waves/block * 32 edges/wave = 256 edges/block -> 1250 blocks exact.
    dgcn_edgeconv_wmma< 64><<<N_EDGES / 256, 256, 0, stream>>>(xbuf, pw0,
                                                               Bs[0], ei);
    dgcn_edgeconv_wmma<128><<<N_EDGES / 256, 256, 0, stream>>>(xbuf, pw1,
                                                               Bs[1], ei);
    dgcn_edgeconv_wmma<192><<<N_EDGES / 256, 256, 0, stream>>>(xbuf, pw2,
                                                               Bs[2], ei);

    dgcn_final_max<<<(N_NODES * 64) / 256, 256, 0, stream>>>(xbuf, out);
}
